// NoteGRU_63522566308355
// MI455X (gfx1250) — compile-verified, matrix operand order checked by reading, not run on hardware
//
#include <hip/hip_runtime.h>

// NoteGRU for MI455X (gfx1250, wave32, WMMA bf16 w/ f32 accum).
//
// Phases:
//  1) convert inputs / h_bar / weights to bf16 in ws
//  2) W_comb = W_ih @ W_in (768x512) + b_comb   (tiny precompute)
//  3) gx      = inputs @ W_comb^T + b_comb      (WMMA GEMM, 32768x512->768)
//     h_reset = h_bar  @ W_init^T + b_init      (WMMA GEMM, 32768x256->256)
//  4) sequential scan: ONE workgroup, 16 waves, all 8 batches as one padded
//     16-row WMMA M-tile; W_hh bf16 fragments VGPR-resident (24 frags/wave);
//     gx[t] AND h_reset[t-1] tiles double-buffered into LDS via async-to-LDS
//     loads, so the steady-state step touches no global memory on its
//     latency-critical path (stores excepted).

#define BATCH  8
#define SEQ    4096
#define DMODEL 512
#define DNOTE  256
#define G3     768   // 3 * DNOTE
#define HBPAD  264   // padded row stride (shorts) for bank-conflict-free ds_read_b128

// CDNA5 async-to-LDS support (guarded so compile never fails)
#if defined(__has_builtin)
#if __has_builtin(__builtin_amdgcn_global_load_async_to_lds_b128)
#define HAVE_ASYNC 1
#endif
#endif
#ifndef HAVE_ASYNC
#define HAVE_ASYNC 0
#endif

typedef __attribute__((ext_vector_type(16))) __bf16 v16bf;
typedef __attribute__((ext_vector_type(8)))  float  v8f;
typedef __attribute__((vector_size(16)))     int    v4i_;

#define AS1 __attribute__((address_space(1)))
#define AS3 __attribute__((address_space(3)))

union Frag16 { v16bf v; uint4 q[2]; };

__device__ __forceinline__ unsigned short f2bf(float f) {
  unsigned int u = __float_as_uint(f);
  u += 0x7FFFu + ((u >> 16) & 1u);           // round-to-nearest-even
  return (unsigned short)(u >> 16);
}

// fast activations: v_exp_f32 + v_rcp_f32 (short serial chains for the scan)
__device__ __forceinline__ float sigmoidf_(float x) {
  return __builtin_amdgcn_rcpf(1.0f + __expf(-x));
}
__device__ __forceinline__ float tanhf_(float x) {
  return 2.0f * __builtin_amdgcn_rcpf(1.0f + __expf(-2.0f * x)) - 1.0f;
}

// ISA 7.12.2 16-bit A/B fragment (16x32 MxK / 32x16 KxN, N-major storage):
// lanes 0-15:  row = lane,    K = k0+[0..7]  then k0+[16..23]
// lanes 16-31: row = lane-16, K = k0+[8..15] then k0+[24..31]
__device__ __forceinline__ v16bf load_frag(const unsigned short* base, int ld,
                                           int row0, int k0, int lane) {
  const int r  = row0 + (lane & 15);
  const int kc = k0 + ((lane & 16) ? 8 : 0);
  const unsigned short* p = base + (size_t)r * ld + kc;
  Frag16 f;
  f.q[0] = *(const uint4*)(p);
  f.q[1] = *(const uint4*)(p + 16);
  return f.v;
}

#if HAVE_ASYNC
__device__ __forceinline__ void async_b128(const float* g, float* l) {
  __builtin_amdgcn_global_load_async_to_lds_b128(
      (AS1 v4i_*)(v4i_*)(void*)g, (AS3 v4i_*)(v4i_*)(void*)l, 0, 0);
}
#endif

// ---------------------------------------------------------------- converts
__global__ void f32_to_bf16_k(const float* __restrict__ src,
                              unsigned short* __restrict__ dst, int n) {
  int i = blockIdx.x * blockDim.x + threadIdx.x;
  if (i < n) dst[i] = f2bf(src[i]);
}

// W_comb[n][k] = sum_j W_ih[n][j] * W_in[j][k];  b_comb[n] = b_ih[n] + W_ih[n].b_in
__global__ void combine_weights_k(const float* __restrict__ W_ih,
                                  const float* __restrict__ W_in,
                                  const float* __restrict__ b_ih,
                                  const float* __restrict__ b_in,
                                  unsigned short* __restrict__ Wc,
                                  float* __restrict__ b_comb) {
  __shared__ float row[DNOTE];
  const int n = blockIdx.x;
  for (int j = threadIdx.x; j < DNOTE; j += blockDim.x) row[j] = W_ih[n * DNOTE + j];
  __syncthreads();
  for (int k = threadIdx.x; k < DMODEL; k += blockDim.x) {
    float s = 0.f;
    #pragma unroll 8
    for (int j = 0; j < DNOTE; ++j) s += row[j] * W_in[j * DMODEL + k];
    Wc[n * DMODEL + k] = f2bf(s);
  }
  if (threadIdx.x == 0) {
    float s = b_ih[n];
    for (int j = 0; j < DNOTE; ++j) s += row[j] * b_in[j];
    b_comb[n] = s;
  }
}

// ------------------------------------------------- generic WMMA GEMM + bias
// C[M x N] (f32) = A[M x K](bf16, row major) @ B[N x K](bf16, N-major)^T + bias
__global__ __launch_bounds__(256)
void gemm_bias_bf16(const unsigned short* __restrict__ A,
                    const unsigned short* __restrict__ B,
                    const float* __restrict__ bias,
                    float* __restrict__ C,
                    int K, int N, int Ntiles, int totalTiles) {
  const int lane = threadIdx.x & 31;
  const int wt   = (int)((blockIdx.x * blockDim.x + threadIdx.x) >> 5);
  if (wt >= totalTiles) return;                       // wave-uniform guard
  const int mt = wt / Ntiles;
  const int nt = wt - mt * Ntiles;
  const int m0 = mt << 4, n0 = nt << 4;

  const float bn = bias[n0 + (lane & 15)];
  v8f acc = {bn, bn, bn, bn, bn, bn, bn, bn};

  for (int k0 = 0; k0 < K; k0 += 32) {
    v16bf a = load_frag(A, K, m0, k0, lane);
    v16bf b = load_frag(B, K, n0, k0, lane);
    acc = __builtin_amdgcn_wmma_f32_16x16x32_bf16(false, a, false, b,
                                                  (short)0, acc, false, false);
  }
  // D layout: lane -> N = n0+(lane&15); elem i -> M = m0 + i + 8*(lane>>4)
  const int n  = n0 + (lane & 15);
  const int mb = m0 + ((lane & 16) ? 8 : 0);
  #pragma unroll
  for (int i = 0; i < 8; ++i)
    C[(size_t)(mb + i) * N + n] = acc[i];
}

// ----------------------------------------------------------------- the scan
__global__ __launch_bounds__(512, 1)
void scan_kernel(const unsigned char* __restrict__ bar_mask,  // (B,S) bool (1B)
                 const float* __restrict__ com_t,             // (B,S,1)
                 const float* __restrict__ gx,                // (B,S,768) f32 ws
                 const float* __restrict__ h_reset,           // (B,S,256) f32 ws
                 const unsigned short* __restrict__ Whh_bf,   // (768,256) bf16
                 const float* __restrict__ b_hh,              // (768,)
                 const float* __restrict__ w_time,            // (256,)
                 const float* __restrict__ b_time,            // (1,)
                 float* __restrict__ out_tc,                  // (B,S)
                 float* __restrict__ out_h) {                 // (B,S,256)
  __shared__ float          gh_s[BATCH][G3];        // 24 KB
  __shared__ float          hf_s[BATCH][DNOTE];     //  8 KB  (f32 h state)
  __shared__ unsigned short hb_s[16][HBPAD];        // 8.25 KB (bf16 h, padded rows)
  __shared__ float          wt_s[DNOTE];
  __shared__ float          tcp[BATCH];
#if HAVE_ASYNC
  __shared__ float          gxbuf[2][BATCH][G3];    // 48 KB double buffer (gx[t])
  __shared__ float          hrbuf[2][BATCH][DNOTE]; // 16 KB double buffer (h_reset[t-1])
#endif

  const int tid  = threadIdx.x;
  const int lane = tid & 31;
  const int wave = tid >> 5;                        // 0..15

  for (int i = tid; i < DNOTE; i += 512) wt_s[i] = w_time[i];
  for (int i = tid; i < 16 * HBPAD; i += 512) (&hb_s[0][0])[i] = 0;
  for (int i = tid; i < BATCH * DNOTE; i += 512) (&hf_s[0][0])[i] = 0.f;
  if (tid < BATCH) tcp[tid] = 0.f;
  const float btime = b_time[0];

  // Persistent W_hh fragments + bias: wave owns N-tiles (3*wave + j), j=0..2
  v16bf wfrag[3][8];
  float bnj[3];
  #pragma unroll
  for (int j = 0; j < 3; ++j) {
    const int n0 = (wave * 3 + j) << 4;
    bnj[j] = b_hh[n0 + (lane & 15)];
    #pragma unroll
    for (int k = 0; k < 8; ++k)
      wfrag[j][k] = load_frag(Whh_bf, DNOTE, n0, k * 32, lane);
  }

#if HAVE_ASYNC
  // Prologue: stage gx[t=0] into buffer 0 (3 x b128 per thread = 24 KB).
  // hrbuf[0] stays unstaged: at t=0 prev_bar==0 so its values are never used.
  {
    #pragma unroll
    for (int i = 0; i < 3; ++i) {
      const int c  = tid + 512 * i;               // 0..1535 b128 chunks
      const int bb = c / 192;                     // batch row (192 chunks each)
      const int xo = (c - bb * 192) * 4;          // float offset in row
      async_b128(gx + ((size_t)bb * SEQ + 0) * G3 + xo, &gxbuf[0][bb][xo]);
    }
  }
#endif
  __syncthreads();

  // per-thread gate/elementwise mapping: 4 consecutive d's of one batch
  const int p0 = tid * 4;
  const int b  = p0 >> 8;
  const int d0 = p0 & 255;

  for (int t = 0; t < SEQ; ++t) {
    const int tm1 = t - (t > 0);                    // safe index (==0 at t==0)
    const int pb  = (t > 0) ? (int)bar_mask[b * SEQ + tm1] : 0;

#if HAVE_ASYNC
    // bar 0: async stages issued last step (gx[t], h_reset[t-1]) have landed
#if defined(__has_builtin) && __has_builtin(__builtin_amdgcn_s_wait_asynccnt)
    __builtin_amdgcn_s_wait_asynccnt(0);
#else
    asm volatile("s_wait_asynccnt 0x0" ::: "memory");
#endif
    __syncthreads();
#endif

    // ---- phase 1: h_in = pb ? h_reset[t-1] : h_prev; publish f32+bf16 copies
    {
#if HAVE_ASYNC
      const float* hr = &hrbuf[t & 1][b][0];        // LDS (staged last step)
#else
      const float* hr = h_reset + ((size_t)b * SEQ + tm1) * DNOTE;
#endif
      float tpart = 0.f;
      #pragma unroll
      for (int i = 0; i < 4; ++i) {
        const int d = d0 + i;
        const float h = pb ? hr[d] : hf_s[b][d];
        hf_s[b][d] = h;
        hb_s[b][d] = f2bf(h);
        tpart += h * wt_s[d];
      }
      atomicAdd(&tcp[b], tpart);                    // ds_add_f32
    }
    __syncthreads();                                // bar A

    // ---- time_center (threads 0..7), uses h_before == h_in
    if (tid < BATCH) {
      const int pbb = (t > 0) ? (int)bar_mask[tid * SEQ + tm1] : 0;
      const float tc = pbb ? com_t[tid * SEQ + tm1]
                           : sigmoidf_(tcp[tid] + btime);
      out_tc[(size_t)tid * SEQ + t] = tc;
      tcp[tid] = 0.f;
    }

    // ---- phase 2: gh = h_in @ W_hh^T + b_hh  (WMMA, weights in VGPRs)
    #pragma unroll
    for (int j = 0; j < 3; ++j) {
      const int n0 = (wave * 3 + j) << 4;
      const float bn = bnj[j];
      v8f acc = {bn, bn, bn, bn, bn, bn, bn, bn};
      #pragma unroll
      for (int k = 0; k < 8; ++k) {
        v16bf a = load_frag(&hb_s[0][0], HBPAD, 0, k * 32, lane);
        acc = __builtin_amdgcn_wmma_f32_16x16x32_bf16(false, a, false, wfrag[j][k],
                                                      (short)0, acc, false, false);
      }
      if (lane < 16) {                              // rows 0..7 are real batches
        const int n = n0 + lane;
        #pragma unroll
        for (int i = 0; i < 8; ++i) gh_s[i][n] = acc[i];
      }
    }
    __syncthreads();                                // bar B

    // ---- phase 3: issue async stage for t+1, then gates, h_new, outputs
#if HAVE_ASYNC
    if (t + 1 < SEQ) {
      const int buf = (t + 1) & 1;
      #pragma unroll
      for (int i = 0; i < 3; ++i) {                 // gx[t+1] -> gxbuf[buf]
        const int c  = tid + 512 * i;
        const int bb = c / 192;
        const int xo = (c - bb * 192) * 4;
        async_b128(gx + ((size_t)bb * SEQ + (t + 1)) * G3 + xo, &gxbuf[buf][bb][xo]);
      }
      {                                             // h_reset[t] -> hrbuf[buf]
        const int bb = tid >> 6;                    // 512 threads = 512 b128 chunks
        const int xo = (tid & 63) * 4;
        async_b128(h_reset + ((size_t)bb * SEQ + t) * DNOTE + xo, &hrbuf[buf][bb][xo]);
      }
    }
#endif
    {
#if HAVE_ASYNC
      const float* gxp = &gxbuf[t & 1][b][0];       // LDS (staged last step)
#else
      const float* gxp = gx + ((size_t)b * SEQ + t) * G3;
      if (t + 1 < SEQ) __builtin_prefetch(gxp + G3, 0, 1);
#endif
      float* outp = out_h + ((size_t)b * SEQ + t) * DNOTE;
      #pragma unroll
      for (int i = 0; i < 4; ++i) {
        const int d = d0 + i;
        const float xr = gxp[d], xz = gxp[DNOTE + d], xn = gxp[2 * DNOTE + d];
        const float gr = gh_s[b][d], gz = gh_s[b][DNOTE + d], gn = gh_s[b][2 * DNOTE + d];
        const float r  = sigmoidf_(xr + gr);
        const float z  = sigmoidf_(xz + gz);
        const float nn = tanhf_(xn + r * gn);
        const float hi = hf_s[b][d];
        const float hn = (1.f - z) * nn + z * hi;
        hf_s[b][d] = hn;
        outp[d] = hn;
      }
    }
    // next iteration's bar0 (wait+barrier) protects all cross-wave LDS reuse.
  }
}

// ---------------------------------------------------------------- launcher
extern "C" void kernel_launch(void* const* d_in, const int* in_sizes, int n_in,
                              void* d_out, int out_size, void* d_ws, size_t ws_size,
                              hipStream_t stream) {
  const float*         inputs   = (const float*)d_in[0];
  const unsigned char* bar_mask = (const unsigned char*)d_in[1]; // jnp.bool_ = 1 byte
  const float*         h_bar    = (const float*)d_in[2];
  const float*         com_t    = (const float*)d_in[3];
  const float*         W_in     = (const float*)d_in[4];
  const float*         b_in     = (const float*)d_in[5];
  const float*         W_init   = (const float*)d_in[6];
  const float*         b_init   = (const float*)d_in[7];
  const float*         W_ih     = (const float*)d_in[8];
  const float*         W_hh     = (const float*)d_in[9];
  const float*         b_ih     = (const float*)d_in[10];
  const float*         b_hh     = (const float*)d_in[11];
  const float*         w_time   = (const float*)d_in[12];
  const float*         b_time   = (const float*)d_in[13];

  char* ws = (char*)d_ws;
  unsigned short* inputs_bf = (unsigned short*)ws; ws += (size_t)BATCH * SEQ * DMODEL * 2;
  unsigned short* hbar_bf   = (unsigned short*)ws; ws += (size_t)BATCH * SEQ * DNOTE * 2;
  unsigned short* Wc_bf     = (unsigned short*)ws; ws += (size_t)G3 * DMODEL * 2;
  unsigned short* Winit_bf  = (unsigned short*)ws; ws += (size_t)DNOTE * DNOTE * 2;
  unsigned short* Whh_bf    = (unsigned short*)ws; ws += (size_t)G3 * DNOTE * 2;
  float*          b_comb    = (float*)ws;          ws += 4096;
  float*          gx        = (float*)ws;          ws += (size_t)BATCH * SEQ * G3 * 4;
  float*          h_reset   = (float*)ws;          // (B,S,256) f32

  float* out_tc = (float*)d_out;
  float* out_h  = out_tc + (size_t)BATCH * SEQ;

  // 1) conversions to bf16
  f32_to_bf16_k<<<(BATCH * SEQ * DMODEL) / 256, 256, 0, stream>>>(inputs, inputs_bf, BATCH * SEQ * DMODEL);
  f32_to_bf16_k<<<(BATCH * SEQ * DNOTE) / 256, 256, 0, stream>>>(h_bar, hbar_bf, BATCH * SEQ * DNOTE);
  f32_to_bf16_k<<<(DNOTE * DNOTE) / 256, 256, 0, stream>>>(W_init, Winit_bf, DNOTE * DNOTE);
  f32_to_bf16_k<<<(G3 * DNOTE) / 256, 256, 0, stream>>>(W_hh, Whh_bf, G3 * DNOTE);

  // 2) folded input projection weight
  combine_weights_k<<<G3, 256, 0, stream>>>(W_ih, W_in, b_ih, b_in, Wc_bf, b_comb);

  // 3) parallel WMMA GEMMs
  {
    const int Mtiles = (BATCH * SEQ) / 16;          // 2048
    const int Ntiles = G3 / 16;                     // 48
    const int total  = Mtiles * Ntiles;             // 98304 wave-tiles
    gemm_bias_bf16<<<total / 8, 256, 0, stream>>>(inputs_bf, Wc_bf, b_comb, gx,
                                                  DMODEL, G3, Ntiles, total);
  }
  {
    const int Mtiles = (BATCH * SEQ) / 16;          // 2048
    const int Ntiles = DNOTE / 16;                  // 16
    const int total  = Mtiles * Ntiles;             // 32768 wave-tiles
    gemm_bias_bf16<<<total / 8, 256, 0, stream>>>(hbar_bf, Winit_bf, b_init, h_reset,
                                                  DNOTE, DNOTE, Ntiles, total);
  }

  // 4) sequential scan: one workgroup, 16 waves
  scan_kernel<<<1, 512, 0, stream>>>(bar_mask, com_t, gx, h_reset, Whh_bf, b_hh,
                                     w_time, b_time, out_tc, out_h);
}
